// SEUNET_23089744183881
// MI455X (gfx1250) — matrix operand order repaired and checked by reading, not compiled, not run on hardware
//
#include <hip/hip_runtime.h>

// SEGNN (2 layers) for MI455X / gfx1250, wave32 + WMMA bf16->f32.
// Per layer:
//   P = x @ Wm1[:64], Q = x @ Wm1[64:128]          (node GEMM, WMMA)
//   edge fused: m1 = swish((P[dst]+Q[src]+amf*w128)*(ea@Vm1)+b1)  -> LDS
//               m2 = swish((m1 @ Wm2)*(ea@Vm2)+b2)                (WMMA)
//               atomicAdd f32 into agg[dst]                       (scatter)
//   u1 = swish(([x|agg] @ Wu1)*(na@Vu1)+b1)        (node GEMM, WMMA)
//   x  = 0.7x + 0.3*((u1 @ Wu2)*(na@Vu2)+b2)       (node GEMM, WMMA)
// Weights are pre-swizzled once into the WMMA B-fragment layout so every
// B operand is a single coalesced 32B vector load per lane.

#define NN 50000
#define EE 800000
#define NODE_TILES 3125   // 50000 / 16
#define EDGE_BLOCKS 6250  // 800000 / (8 waves * 16 edges)

typedef __attribute__((ext_vector_type(16))) __bf16 v16bf;
typedef __attribute__((ext_vector_type(8)))  float  v8f;

union ABu { v16bf v; uint4 u[2]; };
union ABh { v16bf v; __bf16 h[16]; };

__device__ inline v8f v8f_zero() {
  v8f z;
#pragma unroll
  for (int i = 0; i < 8; ++i) z[i] = 0.f;
  return z;
}

__device__ inline float swishf(float v) {
  return v * (1.0f / (1.0f + __expf(-v)));
}

__device__ inline v8f wmma_bf16(v16bf a, v16bf b, v8f c) {
  // (neg_a, A, neg_b, B, c_mod, C, reuse_a, reuse_b)
  return __builtin_amdgcn_wmma_f32_16x16x32_bf16(false, a, false, b, (short)0, c,
                                                 false, false);
}

// A fragment: 16x32 bf16 slab of a row-major [16 x 64] tile (stride 64).
// lane<16: M=lane, K = k0..k0+7 and k0+16..k0+23 ; lane>=16: K shifted +8.
__device__ inline v16bf load_A_bf16(const __bf16* tile, int lane, int ktile) {
  const int m  = lane & 15;
  const int k0 = ktile * 32 + ((lane & 16) ? 8 : 0);
  const __bf16* p = tile + m * 64 + k0;
  ABu r;
  r.u[0] = *(const uint4*)(p);
  r.u[1] = *(const uint4*)(p + 16);
  return r.v;
}

// Same A fragment sourced from a row-major f32 [16 x 64] tile (cvt to bf16).
__device__ inline v16bf load_A_f32(const float* tile, int lane, int ktile) {
  const int m  = lane & 15;
  const int k0 = ktile * 32 + ((lane & 16) ? 8 : 0);
  const float* p = tile + m * 64 + k0;
  float4 f0 = *(const float4*)(p);
  float4 f1 = *(const float4*)(p + 4);
  float4 f2 = *(const float4*)(p + 16);
  float4 f3 = *(const float4*)(p + 20);
  ABh r;
  r.h[0] = (__bf16)f0.x;  r.h[1] = (__bf16)f0.y;  r.h[2]  = (__bf16)f0.z;  r.h[3]  = (__bf16)f0.w;
  r.h[4] = (__bf16)f1.x;  r.h[5] = (__bf16)f1.y;  r.h[6]  = (__bf16)f1.z;  r.h[7]  = (__bf16)f1.w;
  r.h[8] = (__bf16)f2.x;  r.h[9] = (__bf16)f2.y;  r.h[10] = (__bf16)f2.z;  r.h[11] = (__bf16)f2.w;
  r.h[12] = (__bf16)f3.x; r.h[13] = (__bf16)f3.y; r.h[14] = (__bf16)f3.z;  r.h[15] = (__bf16)f3.w;
  return r.v;
}

// B fragment from a pre-swizzled weight buffer: [kt][nt][lane][16 bf16].
// One 32B contiguous vector load per lane (coalesced 1KB per wave).
__device__ inline v16bf load_B_frag(const __bf16* Wf, int lane, int ktile, int ntile) {
  return *(const v16bf*)(Wf + ((((ktile * 4 + ntile) * 32) + lane) << 4));
}

// ---------------------------------------------------------------------------
// Weight conversion f32 -> bf16 with B-fragment swizzle.
// Fragment element e of a [K x 64] weight:
//   frag = e/512, lane = (e%512)/16, i = e%16, kt = frag/4, t = frag%4
//   k = kt*32 + (lane&16 ? 8 : 0) + i + (i&8) ; n = t*16 + (lane&15)
__device__ inline void frag_store(__bf16* dstW, const float* srcW, int e) {
  const int within = e & 511;
  const int lane = within >> 4;
  const int i = within & 15;
  const int frag = e >> 9;
  const int kt = frag >> 2;
  const int t = frag & 3;
  const int k = kt * 32 + ((lane & 16) ? 8 : 0) + i + (i & 8);
  const int n = t * 16 + (lane & 15);
  dstW[e] = (__bf16)srcW[k * 64 + n];
}

__global__ __launch_bounds__(256) void seg_wconv(
    const float* __restrict__ Wm1, const float* __restrict__ Wm2,
    const float* __restrict__ Wu1, const float* __restrict__ Wu2,
    __bf16* __restrict__ wWi, __bf16* __restrict__ wWj, __bf16* __restrict__ wWm2,
    __bf16* __restrict__ wWu1, __bf16* __restrict__ wWu2) {
  const int i = blockIdx.x * 256 + threadIdx.x;
  const int l = i / 24576;
  const int off = i % 24576;
  if (off < 4096) {
    frag_store(wWi + l * 4096, Wm1 + l * 8256, off);
  } else if (off < 8192) {
    frag_store(wWj + l * 4096, Wm1 + l * 8256 + 4096, off - 4096);
  } else if (off < 12288) {
    frag_store(wWm2 + l * 4096, Wm2 + l * 4096, off - 8192);
  } else if (off < 20480) {
    frag_store(wWu1 + l * 8192, Wu1 + l * 8192, off - 12288);
  } else {
    frag_store(wWu2 + l * 4096, Wu2 + l * 4096, off - 20480);
  }
}

// bf16 copy of current x + zero the f32 aggregation buffer.
__global__ __launch_bounds__(256) void seg_prep(
    const float* __restrict__ x, __bf16* __restrict__ xb,
    float* __restrict__ aggf, int n) {
  const int i = blockIdx.x * 256 + threadIdx.x;
  if (i < n) { xb[i] = (__bf16)x[i]; aggf[i] = 0.f; }
}

// ---------------------------------------------------------------------------
// P = xb @ Wi ; Q = xb @ Wj      (both [N,64]@[64,64], bf16 out). LDS-free.
__global__ __launch_bounds__(256) void seg_gemm_pq(
    const __bf16* __restrict__ xb, const __bf16* __restrict__ wWi,
    const __bf16* __restrict__ wWj, __bf16* __restrict__ Pb,
    __bf16* __restrict__ Qb) {
  const int tid = threadIdx.x;
  const int wave = tid >> 5, lane = tid & 31;
  const int tile = blockIdx.x * 8 + wave;
  if (tile >= NODE_TILES) return;

  const __bf16* At = xb + (long long)tile * 1024;
  const v16bf a0 = load_A_bf16(At, lane, 0);
  const v16bf a1 = load_A_bf16(At, lane, 1);
  const int lan15 = lane & 15;
  const int roff = (lane & 16) ? 8 : 0;

#pragma unroll
  for (int half = 0; half < 2; ++half) {
    const __bf16* Wf = half ? wWj : wWi;
    __bf16* out = half ? Qb : Pb;
#pragma unroll
    for (int t = 0; t < 4; ++t) {
      v8f acc = v8f_zero();
      acc = wmma_bf16(a0, load_B_frag(Wf, lane, 0, t), acc);
      acc = wmma_bf16(a1, load_B_frag(Wf, lane, 1, t), acc);
#pragma unroll
      for (int r = 0; r < 8; ++r) {
        const long long node = (long long)tile * 16 + r + roff;
        out[node * 64 + t * 16 + lan15] = (__bf16)acc[r];
      }
    }
  }
}

// ---------------------------------------------------------------------------
// Fused edge kernel: stage1 gate+swish -> LDS m1 -> WMMA m1@Wm2 -> gate+swish
// -> f32 atomic scatter-add. One wave = 16 edges; block = 128 edges.
__global__ __launch_bounds__(256) void seg_edge_fused(
    const __bf16* __restrict__ Pb, const __bf16* __restrict__ Qb,
    const float* __restrict__ edge_attr, const float* __restrict__ amf,
    const int* __restrict__ eidx, const __bf16* __restrict__ Wm2f,
    const float* __restrict__ Vm1l, const float* __restrict__ Vm2l,
    const float* __restrict__ wrow, const float* __restrict__ b1l,
    const float* __restrict__ b2l, float* __restrict__ aggf) {
  __shared__ float sVm1[9 * 64];
  __shared__ float sVm2[9 * 64];
  __shared__ float swrow[64], sb1[64], sb2[64];
  __shared__ __align__(16) __bf16 sM1[8][16 * 64];
  __shared__ float sEA[8][16 * 9];
  __shared__ int sDst[8][16];

  const int tid = threadIdx.x;
  for (int i = tid; i < 576; i += 256) { sVm1[i] = Vm1l[i]; sVm2[i] = Vm2l[i]; }
  if (tid < 64) { swrow[tid] = wrow[tid]; sb1[tid] = b1l[tid]; sb2[tid] = b2l[tid]; }
  __syncthreads();

  const int wave = tid >> 5;
  const int lane = tid & 31;
  const int ebase = blockIdx.x * 128 + wave * 16;

  // ---- phase 1: m1 tile (16 edges x 64 feats) in LDS, bf16 ----
  {
    const int el = lane >> 1;       // edge within wave
    const int hf = lane & 1;        // which 32-feature half
    const int e = ebase + el;
    const int sn = eidx[e];         // src (x_j)
    const int dn = eidx[EE + e];    // dst (x_i, aggregation index)
    float ea[9];
#pragma unroll
    for (int a = 0; a < 9; ++a) ea[a] = edge_attr[e * 9 + a];
    if (hf == 0) {
      sDst[wave][el] = dn;
#pragma unroll
      for (int a = 0; a < 9; ++a) sEA[wave][el * 9 + a] = ea[a];
    }
    const float amfv = amf[e];
    const __bf16* Prow = Pb + (long long)dn * 64;
    const __bf16* Qrow = Qb + (long long)sn * 64;
    const int f0 = hf * 32;
#pragma unroll
    for (int j = 0; j < 4; ++j) {
      const int f = f0 + j * 8;
      union { uint4 u; __bf16 h[8]; } pu, qu, mw;
      pu.u = *(const uint4*)(Prow + f);
      qu.u = *(const uint4*)(Qrow + f);
#pragma unroll
      for (int i = 0; i < 8; ++i) {
        const int fi = f + i;
        float pre = (float)pu.h[i] + (float)qu.h[i] + amfv * swrow[fi];
        float g = 0.f;
#pragma unroll
        for (int a = 0; a < 9; ++a) g += ea[a] * sVm1[a * 64 + fi];
        mw.h[i] = (__bf16)swishf(pre * g + sb1[fi]);
      }
      *(uint4*)(&sM1[wave][el * 64 + f]) = mw.u;  // one ds_store_b128
    }
  }
  // Per-wave LDS region + in-order LDS per wave: no block barrier needed,
  // just forbid compile-time reordering across the phase boundary.
  asm volatile("" ::: "memory");

  // ---- phase 2: m2 = swish((m1 @ Wm2) * g2 + b2) ; scatter-add ----
  const v16bf a0 = load_A_bf16(sM1[wave], lane, 0);
  const v16bf a1 = load_A_bf16(sM1[wave], lane, 1);
  v8f acc[4];
#pragma unroll
  for (int t = 0; t < 4; ++t) {
    acc[t] = v8f_zero();
    acc[t] = wmma_bf16(a0, load_B_frag(Wm2f, lane, 0, t), acc[t]);
    acc[t] = wmma_bf16(a1, load_B_frag(Wm2f, lane, 1, t), acc[t]);
  }
  const int lan15 = lane & 15;
  const int roff = (lane & 16) ? 8 : 0;
  float vc[4][9], bc[4];
#pragma unroll
  for (int t = 0; t < 4; ++t) {
    const int n = t * 16 + lan15;
    bc[t] = sb2[n];
#pragma unroll
    for (int a = 0; a < 9; ++a) vc[t][a] = sVm2[a * 64 + n];
  }
#pragma unroll
  for (int r = 0; r < 8; ++r) {
    const int el = r + roff;
    const int dn = sDst[wave][el];
    float ea[9];
#pragma unroll
    for (int a = 0; a < 9; ++a) ea[a] = sEA[wave][el * 9 + a];
#pragma unroll
    for (int t = 0; t < 4; ++t) {
      float g = 0.f;
#pragma unroll
      for (int a = 0; a < 9; ++a) g += ea[a] * vc[t][a];
      const float m2 = swishf(acc[t][r] * g + bc[t]);
      unsafeAtomicAdd(aggf + (long long)dn * 64 + t * 16 + lan15, m2);
    }
  }
}

// ---------------------------------------------------------------------------
// u1 = swish(([x|agg] @ Wu1) * (node_attr@Vu1) + bu1)   (K = 128). LDS-free.
__global__ __launch_bounds__(256) void seg_gemm_u1(
    const __bf16* __restrict__ xb, const float* __restrict__ aggf,
    const float* __restrict__ node_attr, const __bf16* __restrict__ wWu1,
    const float* __restrict__ Vu1l, const float* __restrict__ bu1l,
    __bf16* __restrict__ u1b) {
  const int tid = threadIdx.x;
  const int wave = tid >> 5, lane = tid & 31;
  const int tile = blockIdx.x * 8 + wave;
  if (tile >= NODE_TILES) return;

  v16bf a[4];
  a[0] = load_A_bf16(xb + (long long)tile * 1024, lane, 0);
  a[1] = load_A_bf16(xb + (long long)tile * 1024, lane, 1);
  a[2] = load_A_f32(aggf + (long long)tile * 1024, lane, 0);
  a[3] = load_A_f32(aggf + (long long)tile * 1024, lane, 1);

  v8f acc[4];
#pragma unroll
  for (int t = 0; t < 4; ++t) {
    acc[t] = v8f_zero();
#pragma unroll
    for (int kt = 0; kt < 4; ++kt)
      acc[t] = wmma_bf16(a[kt], load_B_frag(wWu1, lane, kt, t), acc[t]);
  }

  const int lan15 = lane & 15;
  const int roff = (lane & 16) ? 8 : 0;
  float vc[4][9], bc[4];
#pragma unroll
  for (int t = 0; t < 4; ++t) {
    const int n = t * 16 + lan15;
    bc[t] = bu1l[n];
#pragma unroll
    for (int a2 = 0; a2 < 9; ++a2) vc[t][a2] = Vu1l[a2 * 64 + n];
  }
#pragma unroll
  for (int r = 0; r < 8; ++r) {
    const long long node = (long long)tile * 16 + r + roff;
    float na[9];
#pragma unroll
    for (int a2 = 0; a2 < 9; ++a2) na[a2] = node_attr[node * 9 + a2];
#pragma unroll
    for (int t = 0; t < 4; ++t) {
      float g = 0.f;
#pragma unroll
      for (int a2 = 0; a2 < 9; ++a2) g += na[a2] * vc[t][a2];
      u1b[node * 64 + t * 16 + lan15] = (__bf16)swishf(acc[t][r] * g + bc[t]);
    }
  }
}

// ---------------------------------------------------------------------------
// x_out = 0.7*x_in + 0.3*((u1 @ Wu2) * (node_attr@Vu2) + bu2). LDS-free.
__global__ __launch_bounds__(256) void seg_gemm_u2(
    const __bf16* __restrict__ u1b, const float* __restrict__ xin,
    const float* __restrict__ node_attr, const __bf16* __restrict__ wWu2,
    const float* __restrict__ Vu2l, const float* __restrict__ bu2l,
    float* __restrict__ xout) {
  const int tid = threadIdx.x;
  const int wave = tid >> 5, lane = tid & 31;
  const int tile = blockIdx.x * 8 + wave;
  if (tile >= NODE_TILES) return;

  const v16bf a0 = load_A_bf16(u1b + (long long)tile * 1024, lane, 0);
  const v16bf a1 = load_A_bf16(u1b + (long long)tile * 1024, lane, 1);
  v8f acc[4];
#pragma unroll
  for (int t = 0; t < 4; ++t) {
    acc[t] = v8f_zero();
    acc[t] = wmma_bf16(a0, load_B_frag(wWu2, lane, 0, t), acc[t]);
    acc[t] = wmma_bf16(a1, load_B_frag(wWu2, lane, 1, t), acc[t]);
  }

  const int lan15 = lane & 15;
  const int roff = (lane & 16) ? 8 : 0;
  float vc[4][9], bc[4];
#pragma unroll
  for (int t = 0; t < 4; ++t) {
    const int n = t * 16 + lan15;
    bc[t] = bu2l[n];
#pragma unroll
    for (int a2 = 0; a2 < 9; ++a2) vc[t][a2] = Vu2l[a2 * 64 + n];
  }
#pragma unroll
  for (int r = 0; r < 8; ++r) {
    const long long node = (long long)tile * 16 + r + roff;
    float na[9];
#pragma unroll
    for (int a2 = 0; a2 < 9; ++a2) na[a2] = node_attr[node * 9 + a2];
#pragma unroll
    for (int t = 0; t < 4; ++t) {
      float g = 0.f;
#pragma unroll
      for (int a2 = 0; a2 < 9; ++a2) g += na[a2] * vc[t][a2];
      const float u = acc[t][r] * g + bc[t];
      const long long idx = node * 64 + t * 16 + lan15;
      xout[idx] = 0.7f * xin[idx] + 0.3f * u;
    }
  }
}

// ---------------------------------------------------------------------------
extern "C" void kernel_launch(void* const* d_in, const int* in_sizes, int n_in,
                              void* d_out, int out_size, void* d_ws, size_t ws_size,
                              hipStream_t stream) {
  (void)in_sizes; (void)n_in; (void)out_size; (void)ws_size;

  const float* x         = (const float*)d_in[0];
  const float* edge_attr = (const float*)d_in[1];
  const float* node_attr = (const float*)d_in[2];
  const float* amf       = (const float*)d_in[3];
  const float* Wm1       = (const float*)d_in[4];
  const float* Vm1       = (const float*)d_in[5];
  const float* bm1       = (const float*)d_in[6];
  const float* Wm2       = (const float*)d_in[7];
  const float* Vm2       = (const float*)d_in[8];
  const float* bm2       = (const float*)d_in[9];
  const float* Wu1       = (const float*)d_in[10];
  const float* Vu1       = (const float*)d_in[11];
  const float* bu1       = (const float*)d_in[12];
  const float* Wu2       = (const float*)d_in[13];
  const float* Vu2       = (const float*)d_in[14];
  const float* bu2       = (const float*)d_in[15];
  const int*   eidx      = (const int*)d_in[16];
  float* out = (float*)d_out;

  char* wsp = (char*)d_ws;
  auto carve = [&](size_t bytes) {
    char* p = wsp;
    wsp += (bytes + 255) & ~(size_t)255;
    return p;
  };
  __bf16* wWi  = (__bf16*)carve(2 * 4096 * sizeof(__bf16));
  __bf16* wWj  = (__bf16*)carve(2 * 4096 * sizeof(__bf16));
  __bf16* wWm2 = (__bf16*)carve(2 * 4096 * sizeof(__bf16));
  __bf16* wWu1 = (__bf16*)carve(2 * 8192 * sizeof(__bf16));
  __bf16* wWu2 = (__bf16*)carve(2 * 4096 * sizeof(__bf16));
  __bf16* xb   = (__bf16*)carve((size_t)NN * 64 * sizeof(__bf16));
  __bf16* Pb   = (__bf16*)carve((size_t)NN * 64 * sizeof(__bf16));
  __bf16* Qb   = (__bf16*)carve((size_t)NN * 64 * sizeof(__bf16));
  __bf16* u1b  = (__bf16*)carve((size_t)NN * 64 * sizeof(__bf16));
  float*  aggf = (float*)carve((size_t)NN * 64 * sizeof(float));
  float*  xbuf = (float*)carve((size_t)NN * 64 * sizeof(float));

  seg_wconv<<<192, 256, 0, stream>>>(Wm1, Wm2, Wu1, Wu2, wWi, wWj, wWm2, wWu1, wWu2);

  const int nElem = NN * 64;
  const int prepBlocks = (nElem + 255) / 256;
  const int nodeBlocks = (NODE_TILES + 7) / 8;

  for (int l = 0; l < 2; ++l) {
    const float* xcur = (l == 0) ? x : xbuf;
    float* xout = (l == 0) ? xbuf : out;

    seg_prep<<<prepBlocks, 256, 0, stream>>>(xcur, xb, aggf, nElem);

    seg_gemm_pq<<<nodeBlocks, 256, 0, stream>>>(xb, wWi + l * 4096, wWj + l * 4096,
                                                Pb, Qb);

    seg_edge_fused<<<EDGE_BLOCKS, 256, 0, stream>>>(
        Pb, Qb, edge_attr, amf, eidx, wWm2 + l * 4096, Vm1 + l * 576,
        Vm2 + l * 576, Wm1 + l * 8256 + 8192, bm1 + l * 64, bm2 + l * 64, aggf);

    seg_gemm_u1<<<nodeBlocks, 256, 0, stream>>>(xb, aggf, node_attr,
                                                wWu1 + l * 8192, Vu1 + l * 576,
                                                bu1 + l * 64, u1b);

    seg_gemm_u2<<<nodeBlocks, 256, 0, stream>>>(u1b, xcur, node_attr,
                                                wWu2 + l * 4096, Vu2 + l * 576,
                                                bu2 + l * 64, xout);
  }
}
